// MMK_Mix_3968549781577
// MI455X (gfx1250) — matrix operand order repaired and checked by reading, not compiled, not run on hardware
//
#include <hip/hip_runtime.h>
#include <hip/hip_bf16.h>
#include <math.h>

// ---------------------------------------------------------------------------
// MoK-KAN forecaster for MI455X (gfx1250), compute-bound on the fused
// spline+base expert GEMM (2048 x 4608) x (4608 x E*out_d).
// - v_wmma_f32_16x16x32_bf16, wave tile 32x64 (8 WMMA / K-step), block 128x128
// - expert GEMM tiles staged LDS-side by the Tensor Data Mover
//   (tensor_load_to_lds + s_wait_tensorcnt), gate GEMM uses manual staging
//   with global_prefetch_b8.
// ---------------------------------------------------------------------------

typedef __bf16 bf16_t;
typedef __attribute__((ext_vector_type(16))) __bf16 v16bf;
typedef __attribute__((ext_vector_type(8)))  float  v8f;
typedef unsigned int v4u __attribute__((ext_vector_type(4)));
typedef unsigned int v8u __attribute__((ext_vector_type(8)));

struct alignas(16) U4 { unsigned int a, b, c, d; };
struct alignas(16) F4 { float x, y, z, w; };

#define EPSF 1e-5f

static constexpr int Bx = 64, Lx = 512, Nvar = 32, HIDx = 512, PREDx = 96;
static constexpr int Ex = 8, Gx = 8;
static constexpr int Mrows = Bx * Nvar;          // 2048
static constexpr int KBIG  = HIDx * Gx + HIDx;   // 4608 (basis + silu branch)

// ---------------------------------------------------------------------------
// Tensor Data Mover: async-load a (128 rows x 32 col) bf16 tile into LDS.
// D# group0/group1 packed per CDNA5 ISA 8.3/8.4; groups 2,3 zero (2D tile).
// Issue from one wave only; completion via s_wait_tensorcnt.
// ---------------------------------------------------------------------------
__device__ __forceinline__ void tdm_load_tile_128x32(
    unsigned lds_off, const bf16_t* gptr, unsigned tensor_rows,
    unsigned tensor_cols, unsigned long long stride_elems)
{
  const unsigned long long ga = (unsigned long long)(uintptr_t)gptr;
  v4u g0;
  g0.x = 1u;                                    // count=1, user descriptor
  g0.y = lds_off;                               // LDS byte address
  g0.z = (unsigned)ga;                          // global_addr[31:0]
  g0.w = ((unsigned)(ga >> 32) & 0x1ffffffu)    // global_addr[56:32]
         | (2u << 30);                          // type=2 ("image")
  v8u g1;
  g1[0] = 1u << 16;                             // data_size=1 -> 2 bytes
  g1[1] = (tensor_cols & 0xffffu) << 16;        // tensor_dim0[15:0]
  g1[2] = (tensor_cols >> 16)                   // tensor_dim0[31:16]
          | ((tensor_rows & 0xffffu) << 16);    // tensor_dim1[15:0]
  g1[3] = (tensor_rows >> 16)                   // tensor_dim1[31:16]
          | (32u << 16);                        // tile_dim0 = 32 elements
  g1[4] = 128u;                                 // tile_dim1 = 128, tile_dim2=0
  g1[5] = (unsigned)stride_elems;               // tensor_dim0_stride[31:0]
  g1[6] = (unsigned)(stride_elems >> 32);       // stride[47:32]; dim1_stride=0
  g1[7] = 0u;
  const v4u gz = {0u, 0u, 0u, 0u};              // groups 2/3 unused (2D)
  asm volatile("tensor_load_to_lds %0, %1, %2, %3"
               :: "s"(g0), "s"(g1), "s"(gz), "s"(gz)
               : "memory");
}

// ---------------------------------------------------------------------------
// RevIN: per (b,n) series: mean/std over time, normalize, transpose to
// x[(b*N+n), l].
// ---------------------------------------------------------------------------
__global__ __launch_bounds__(256) void revin_kernel(
    const float* __restrict__ vx, const float* __restrict__ gam,
    const float* __restrict__ bet, float* __restrict__ x,
    float* __restrict__ meanv, float* __restrict__ stdv)
{
  const int row = blockIdx.x;                // 0..2047
  const int b = row >> 5, n = row & 31;
  __shared__ float ss[256], sq[256];
  float s = 0.f, q = 0.f;
  for (int l = threadIdx.x; l < Lx; l += 256) {
    float v = vx[((size_t)b * Lx + l) * Nvar + n];
    s += v; q += v * v;
  }
  ss[threadIdx.x] = s; sq[threadIdx.x] = q;
  __syncthreads();
  for (int st = 128; st > 0; st >>= 1) {
    if (threadIdx.x < st) {
      ss[threadIdx.x] += ss[threadIdx.x + st];
      sq[threadIdx.x] += sq[threadIdx.x + st];
    }
    __syncthreads();
  }
  const float mu  = ss[0] * (1.f / Lx);
  const float var = sq[0] * (1.f / Lx) - mu * mu;
  const float sd  = sqrtf(var + EPSF);
  if (threadIdx.x == 0) { meanv[row] = mu; stdv[row] = sd; }
  const float ga = gam[n], be = bet[n];
  for (int l = threadIdx.x; l < Lx; l += 256) {
    float v = vx[((size_t)b * Lx + l) * Nvar + n];
    x[(size_t)row * Lx + l] = (v - mu) / sd * ga + be;
  }
}

// ---------------------------------------------------------------------------
// Gate GEMM: C_bf16[m,n] = relu(sum_k A_f32[m,k]*B_bf16[n,k] + bias[n])
// Block tile 128x128, wave tile 32x64, manual staging with fp32->bf16 cvt.
// ---------------------------------------------------------------------------
__global__ __launch_bounds__(256) void gemm_gate_kernel(
    const float* __restrict__ A, const bf16_t* __restrict__ Bmat,
    bf16_t* __restrict__ C, const float* __restrict__ bias,
    int K, int lda, int ldc)
{
  __shared__ bf16_t As[128 * 32];
  __shared__ bf16_t Bs[128 * 32];

  const int tid  = threadIdx.x;
  const int m0   = blockIdx.x * 128;
  const int n0   = blockIdx.y * 128;
  const int wave = tid >> 5;
  const int lane = tid & 31;
  const int wm   = (wave & 3) * 32;
  const int wn   = (wave >> 2) * 64;
  const int lm   = lane & 15;
  const int hi   = lane >> 4;

  v8f acc[2][4] = {};

  for (int k0 = 0; k0 < K; k0 += 32) {
    #pragma unroll
    for (int it = 0; it < 2; ++it) {
      const int i = tid + it * 256;            // 512 groups of 8
      const int m = i >> 2;
      const int k = (i & 3) << 3;
      const float* src = &A[(size_t)(m0 + m) * lda + (k0 + k)];
      const F4 f0 = *(const F4*)src;
      const F4 f1 = *(const F4*)(src + 4);
      union { bf16_t h[8]; U4 u; } t;
      t.h[0] = (bf16_t)f0.x; t.h[1] = (bf16_t)f0.y;
      t.h[2] = (bf16_t)f0.z; t.h[3] = (bf16_t)f0.w;
      t.h[4] = (bf16_t)f1.x; t.h[5] = (bf16_t)f1.y;
      t.h[6] = (bf16_t)f1.z; t.h[7] = (bf16_t)f1.w;
      *(U4*)&As[m * 32 + k] = t.u;
      *(U4*)&Bs[m * 32 + k] =
          *(const U4*)&Bmat[(size_t)(n0 + m) * K + (k0 + k)];
      if (k0 + 32 < K)
        __builtin_prefetch(&Bmat[(size_t)(n0 + m) * K + (k0 + 32 + k)], 0, 1);
    }
    __syncthreads();

    union Frag { U4 u[2]; v16bf v; };
    Frag afr[2], bfr[4];
    #pragma unroll
    for (int im = 0; im < 2; ++im) {
      const bf16_t* p = &As[(wm + im * 16 + lm) * 32];
      afr[im].u[0] = *(const U4*)&p[hi * 8];
      afr[im].u[1] = *(const U4*)&p[16 + hi * 8];
    }
    #pragma unroll
    for (int jn = 0; jn < 4; ++jn) {
      const bf16_t* p = &Bs[(wn + jn * 16 + lm) * 32];
      bfr[jn].u[0] = *(const U4*)&p[hi * 8];
      bfr[jn].u[1] = *(const U4*)&p[16 + hi * 8];
    }
    #pragma unroll
    for (int im = 0; im < 2; ++im)
      #pragma unroll
      for (int jn = 0; jn < 4; ++jn)
        acc[im][jn] = __builtin_amdgcn_wmma_f32_16x16x32_bf16(
            false, afr[im].v, false, bfr[jn].v, (short)0, acc[im][jn],
            false, false);
    __syncthreads();
  }

  #pragma unroll
  for (int im = 0; im < 2; ++im) {
    #pragma unroll
    for (int jn = 0; jn < 4; ++jn) {
      const int col   = n0 + wn + jn * 16 + lm;
      const int rbase = m0 + wm + im * 16 + hi * 8;
      const float bcol = bias[col];
      #pragma unroll
      for (int v = 0; v < 8; ++v) {
        float val = acc[im][jn][v] + bcol;
        val = val > 0.f ? val : 0.f;
        C[(size_t)(rbase + v) * ldc + col] = (bf16_t)val;
      }
    }
  }
}

// ---------------------------------------------------------------------------
// Expert GEMM: C_f32[m,n] = sum_k A_bf16[m,k]*B_bf16[n,k]
// A/B tiles staged by the Tensor Data Mover (async DMA into LDS).
// ---------------------------------------------------------------------------
__global__ __launch_bounds__(256) void gemm_expert_kernel(
    const bf16_t* __restrict__ A, const bf16_t* __restrict__ Bmat,
    float* __restrict__ C, int M, int N, int K, int lda, int ldc)
{
  __shared__ bf16_t As[128 * 32];
  __shared__ bf16_t Bs[128 * 32];

  const int tid  = threadIdx.x;
  const int m0   = blockIdx.x * 128;
  const int n0   = blockIdx.y * 128;
  const int wave = tid >> 5;
  const int lane = tid & 31;
  const int wm   = (wave & 3) * 32;
  const int wn   = (wave >> 2) * 64;
  const int lm   = lane & 15;
  const int hi   = lane >> 4;

  // low 32 bits of the flat shared address == LDS byte offset
  const unsigned ldsA = (unsigned)(uintptr_t)&As[0];
  const unsigned ldsB = (unsigned)(uintptr_t)&Bs[0];

  v8f acc[2][4] = {};

  for (int k0 = 0; k0 < K; k0 += 32) {
    if (wave == 0) {
      tdm_load_tile_128x32(ldsA, &A[(size_t)m0 * lda + k0],
                           (unsigned)M, (unsigned)K,
                           (unsigned long long)lda);
      tdm_load_tile_128x32(ldsB, &Bmat[(size_t)n0 * (size_t)K + k0],
                           (unsigned)N, (unsigned)K,
                           (unsigned long long)K);
      __builtin_amdgcn_s_wait_tensorcnt(0);
    }
    __syncthreads();

    union Frag { U4 u[2]; v16bf v; };
    Frag afr[2], bfr[4];
    #pragma unroll
    for (int im = 0; im < 2; ++im) {
      const bf16_t* p = &As[(wm + im * 16 + lm) * 32];
      afr[im].u[0] = *(const U4*)&p[hi * 8];         // K = hi*8 .. +7
      afr[im].u[1] = *(const U4*)&p[16 + hi * 8];    // K = 16+hi*8 .. +7
    }
    #pragma unroll
    for (int jn = 0; jn < 4; ++jn) {
      const bf16_t* p = &Bs[(wn + jn * 16 + lm) * 32];
      bfr[jn].u[0] = *(const U4*)&p[hi * 8];         // K = hi*8 .. +7
      bfr[jn].u[1] = *(const U4*)&p[16 + hi * 8];    // K = 16+hi*8 .. +7
    }
    #pragma unroll
    for (int im = 0; im < 2; ++im)
      #pragma unroll
      for (int jn = 0; jn < 4; ++jn)
        acc[im][jn] = __builtin_amdgcn_wmma_f32_16x16x32_bf16(
            false, afr[im].v, false, bfr[jn].v, (short)0, acc[im][jn],
            false, false);
    __syncthreads();
  }

  #pragma unroll
  for (int im = 0; im < 2; ++im) {
    #pragma unroll
    for (int jn = 0; jn < 4; ++jn) {
      const int col   = n0 + wn + jn * 16 + lm;
      const int rbase = m0 + wm + im * 16 + hi * 8;
      #pragma unroll
      for (int v = 0; v < 8; ++v)
        C[(size_t)(rbase + v) * ldc + col] = acc[im][jn][v];
    }
  }
}

// ---------------------------------------------------------------------------
// Gate second layer + softmax.
// ---------------------------------------------------------------------------
__global__ __launch_bounds__(256) void gate2_softmax_kernel(
    const bf16_t* __restrict__ g, const float* __restrict__ gW2,
    const float* __restrict__ gb2, float* __restrict__ scores)
{
  const int row = blockIdx.x * 256 + threadIdx.x;
  if (row >= Mrows) return;
  float s[Ex];
  #pragma unroll
  for (int e = 0; e < Ex; ++e) s[e] = gb2[e];
  const bf16_t* gr = &g[(size_t)row * HIDx];
  for (int k = 0; k < HIDx; ++k) {
    const float gv = (float)gr[k];
    #pragma unroll
    for (int e = 0; e < Ex; ++e) s[e] += gv * gW2[e * HIDx + k];
  }
  float mx = s[0];
  #pragma unroll
  for (int e = 1; e < Ex; ++e) mx = fmaxf(mx, s[e]);
  float sum = 0.f;
  #pragma unroll
  for (int e = 0; e < Ex; ++e) { s[e] = __expf(s[e] - mx); sum += s[e]; }
  const float inv = 1.f / sum;
  #pragma unroll
  for (int e = 0; e < Ex; ++e) scores[(size_t)row * Ex + e] = s[e] * inv;
}

// ---------------------------------------------------------------------------
// Expand x -> A_big bf16 (2048 x 4608): RBF basis || silu(x).
// ---------------------------------------------------------------------------
__global__ __launch_bounds__(256) void expand_basis_kernel(
    const float* __restrict__ x, bf16_t* __restrict__ A)
{
  const int idx = blockIdx.x * 256 + threadIdx.x;   // 2048*512
  const int m = idx >> 9;
  const int i = idx & 511;
  const float v = x[idx];
  union { bf16_t h[8]; U4 u; } t;
  #pragma unroll
  for (int gg = 0; gg < Gx; ++gg) {
    const float c  = -2.f + (4.f / 7.f) * gg;
    const float tt = (v - c) * 1.75f;
    t.h[gg] = (bf16_t)__expf(-tt * tt);
  }
  *(U4*)&A[(size_t)m * KBIG + i * 8] = t.u;
  A[(size_t)m * KBIG + 4096 + i] = (bf16_t)(v / (1.f + __expf(-v)));
}

// Pack [Ws | Wb] -> bf16, row c=(e*out_d+o): 4096 spline + 512 base weights.
__global__ __launch_bounds__(256) void pack_wcomb_kernel(
    const float* __restrict__ Ws, const float* __restrict__ Wb,
    bf16_t* __restrict__ W, int out_d)
{
  const size_t total = (size_t)Ex * out_d * KBIG;
  for (size_t idx = (size_t)blockIdx.x * 256 + threadIdx.x; idx < total;
       idx += (size_t)gridDim.x * 256) {
    const size_t c = idx / KBIG;
    const int k = (int)(idx - c * KBIG);
    const float v = (k < 4096) ? Ws[c * 4096 + k] : Wb[c * 512 + (k - 4096)];
    W[idx] = (bf16_t)v;
  }
}

// gW1 (K x N) row-major -> bf16 (N x K) so GEMM B rows are K-contiguous.
__global__ __launch_bounds__(256) void pack_gw1_kernel(
    const float* __restrict__ gW1, bf16_t* __restrict__ W)
{
  const int idx = blockIdx.x * 256 + threadIdx.x;   // 512*512
  const int n = idx >> 9, k = idx & 511;
  W[idx] = (bf16_t)gW1[k * 512 + n];
}

// y[m,o] = sum_e scores[m,e]*(EO[m,e*out_d+o]+b[e,o]) (+ residual x[m,o])
__global__ __launch_bounds__(256) void reduce_experts_kernel(
    const float* __restrict__ EO, const float* __restrict__ scores,
    const float* __restrict__ bexp, const float* __restrict__ xin,
    float* __restrict__ y, int out_d, int res)
{
  const int idx = blockIdx.x * 256 + threadIdx.x;
  const int m = idx / out_d;
  const int o = idx - m * out_d;
  if (m >= Mrows) return;
  const int Ncols = Ex * out_d;
  float a = 0.f;
  #pragma unroll
  for (int e = 0; e < Ex; ++e)
    a += scores[(size_t)m * Ex + e] *
         (EO[(size_t)m * Ncols + e * out_d + o] + bexp[e * out_d + o]);
  if (res) a += xin[(size_t)m * HIDx + o];
  y[(size_t)m * out_d + o] = a;
}

// Training-mode BatchNorm over the 2048-row batch dim.
__global__ __launch_bounds__(256) void bn_stats_kernel(
    const float* __restrict__ y, float* __restrict__ mu,
    float* __restrict__ rs, int out_d)
{
  const int o = blockIdx.x;
  __shared__ float ss[256], sq[256];
  float s = 0.f, q = 0.f;
  for (int m = threadIdx.x; m < Mrows; m += 256) {
    const float v = y[(size_t)m * out_d + o];
    s += v; q += v * v;
  }
  ss[threadIdx.x] = s; sq[threadIdx.x] = q;
  __syncthreads();
  for (int st = 128; st > 0; st >>= 1) {
    if (threadIdx.x < st) {
      ss[threadIdx.x] += ss[threadIdx.x + st];
      sq[threadIdx.x] += sq[threadIdx.x + st];
    }
    __syncthreads();
  }
  if (threadIdx.x == 0) {
    const float mean = ss[0] * (1.f / Mrows);
    const float var  = sq[0] * (1.f / Mrows) - mean * mean;
    mu[o] = mean;
    rs[o] = rsqrtf(var + EPSF);
  }
}

__global__ __launch_bounds__(256) void bn_apply_kernel(
    float* __restrict__ y, const float* __restrict__ mu,
    const float* __restrict__ rs, const float* __restrict__ gam,
    const float* __restrict__ bet, int out_d)
{
  const int idx = blockIdx.x * 256 + threadIdx.x;
  const int m = idx / out_d;
  const int o = idx - m * out_d;
  if (m >= Mrows) return;
  y[idx] = (y[idx] - mu[o]) * rs[o] * gam[o] + bet[o];
}

// out[b,t,n] = (y3[b*N+n, t] - rbeta[n]) / max(rgamma[n],eps) * std + mean
__global__ __launch_bounds__(256) void final_denorm_kernel(
    const float* __restrict__ y3, const float* __restrict__ gam,
    const float* __restrict__ bet, const float* __restrict__ meanv,
    const float* __restrict__ stdv, float* __restrict__ out)
{
  const int idx = blockIdx.x * 256 + threadIdx.x;   // 64*96*32
  const int n = idx & 31;
  const int t = (idx >> 5) % PREDx;
  const int b = idx / (PREDx * Nvar);
  const int row = b * Nvar + n;
  const float gc = fmaxf(gam[n], EPSF);
  out[idx] = (y3[(size_t)row * PREDx + t] - bet[n]) / gc * stdv[row]
             + meanv[row];
}

// ---------------------------------------------------------------------------
extern "C" void kernel_launch(void* const* d_in, const int* in_sizes, int n_in,
                              void* d_out, int out_size, void* d_ws,
                              size_t ws_size, hipStream_t stream)
{
  (void)in_sizes; (void)n_in; (void)out_size; (void)ws_size;
  // d_in order: var_x, then per layer {Ws,Wb,b,gW1,gb1,gW2,gb2,bn_g,bn_b} x3,
  // then revin_gamma, revin_beta.
  const float* var_x = (const float*)d_in[0];
  const float* rgam  = (const float*)d_in[28];
  const float* rbet  = (const float*)d_in[29];

  char* w = (char*)d_ws;
  auto carve = [&](size_t bytes) -> void* {
    void* p = (void*)w;
    w += (bytes + 255) & ~(size_t)255;
    return p;
  };
  float*  xA     = (float*) carve((size_t)Mrows * HIDx * 4);
  float*  xB     = (float*) carve((size_t)Mrows * HIDx * 4);
  float*  meanv  = (float*) carve((size_t)Mrows * 4);
  float*  stdv   = (float*) carve((size_t)Mrows * 4);
  bf16_t* ghid   = (bf16_t*)carve((size_t)Mrows * HIDx * 2);
  float*  scores = (float*) carve((size_t)Mrows * Ex * 4);
  bf16_t* Abig   = (bf16_t*)carve((size_t)Mrows * KBIG * 2);
  float*  EO     = (float*) carve((size_t)Mrows * Ex * HIDx * 4);
  bf16_t* Wcomb  = (bf16_t*)carve((size_t)Ex * HIDx * KBIG * 2);
  bf16_t* gW1bf  = (bf16_t*)carve((size_t)HIDx * HIDx * 2);
  float*  bn_mu  = (float*) carve(HIDx * 4);
  float*  bn_rs  = (float*) carve(HIDx * 4);

  revin_kernel<<<Mrows, 256, 0, stream>>>(var_x, rgam, rbet, xA, meanv, stdv);

  float* xcur = xA;
  float* xnext = xB;
  for (int l = 0; l < 3; ++l) {
    const float* Ws  = (const float*)d_in[1 + l * 9 + 0];
    const float* Wb  = (const float*)d_in[1 + l * 9 + 1];
    const float* bb  = (const float*)d_in[1 + l * 9 + 2];
    const float* gW1 = (const float*)d_in[1 + l * 9 + 3];
    const float* gb1 = (const float*)d_in[1 + l * 9 + 4];
    const float* gW2 = (const float*)d_in[1 + l * 9 + 5];
    const float* gb2 = (const float*)d_in[1 + l * 9 + 6];
    const float* bng = (const float*)d_in[1 + l * 9 + 7];
    const float* bnb = (const float*)d_in[1 + l * 9 + 8];
    const int out_d = (l == 2) ? PREDx : HIDx;
    const int Ncols = Ex * out_d;

    // gate: hidden = relu(x @ gW1 + gb1)  [WMMA, fp32 A -> bf16 out]
    pack_gw1_kernel<<<(HIDx * HIDx) / 256, 256, 0, stream>>>(gW1, gW1bf);
    gemm_gate_kernel<<<dim3(Mrows / 128, HIDx / 128), 256, 0, stream>>>(
        xcur, gW1bf, ghid, gb1, HIDx, HIDx, HIDx);
    gate2_softmax_kernel<<<Mrows / 256, 256, 0, stream>>>(ghid, gW2, gb2,
                                                          scores);

    // fused spline+base expert GEMM (TDM-staged tiles)
    expand_basis_kernel<<<(Mrows * HIDx) / 256, 256, 0, stream>>>(xcur, Abig);
    pack_wcomb_kernel<<<4096, 256, 0, stream>>>(Ws, Wb, Wcomb, out_d);
    gemm_expert_kernel<<<dim3(Mrows / 128, Ncols / 128), 256, 0, stream>>>(
        Abig, Wcomb, EO, Mrows, Ncols, KBIG, KBIG, Ncols);

    const int res = (out_d == HIDx) ? 1 : 0;
    reduce_experts_kernel<<<(Mrows * out_d) / 256, 256, 0, stream>>>(
        EO, scores, bb, xcur, xnext, out_d, res);
    bn_stats_kernel<<<out_d, 256, 0, stream>>>(xnext, bn_mu, bn_rs, out_d);
    bn_apply_kernel<<<(Mrows * out_d) / 256, 256, 0, stream>>>(
        xnext, bn_mu, bn_rs, bng, bnb, out_d);

    float* t = xcur; xcur = xnext; xnext = t;
  }

  final_denorm_kernel<<<(Bx * PREDx * Nvar) / 256, 256, 0, stream>>>(
      xcur, rgam, rbet, meanv, stdv, (float*)d_out);
}
